// UpDownDecoder_7748121002689
// MI455X (gfx1250) — matrix operand order repaired and checked by reading, not compiled, not run on hardware
//
#include <hip/hip_runtime.h>
#include <hip/hip_bf16.h>
#include <cstdint>
#include <cstddef>

// ---------------------------------------------------------------------------
// Problem constants (from the reference): V,E,H,D,R fixed; B and steps derived
// ---------------------------------------------------------------------------
constexpr int V_  = 10000;
constexpr int VP_ = 10048;      // vocab padded to a multiple of 64 (157 strips)
constexpr int E_  = 256;
constexpr int H_  = 512;
constexpr int D_  = 2048;
constexpr int R_  = 36;
constexpr int EH_ = E_ + H_;    // 768  (LSTM input x = [emb | ctx])
constexpr int XH_ = EH_ + H_;   // 1280 (packed [emb | ctx | h] row)
constexpr int G4_ = 4 * H_;     // 2048 gates

// ---------------------------------------------------------------------------
// WMMA types (CDNA5 gfx1250, wave32)
// ---------------------------------------------------------------------------
typedef __attribute__((ext_vector_type(16))) __bf16 v16bf;
typedef __attribute__((ext_vector_type(8)))  __bf16 v8bf;
typedef __attribute__((ext_vector_type(8)))  float  v8f;

union ABfrag { v16bf v; v8bf p[2]; };

__device__ __forceinline__ v16bf load_frag16(const __bf16* p0, const __bf16* p1) {
  ABfrag f;
  f.p[0] = *(const v8bf*)p0;
  f.p[1] = *(const v8bf*)p1;
  return f.v;
}

__device__ __forceinline__ v8f wmma_bf16(v16bf a, v16bf b, v8f c) {
  return __builtin_amdgcn_wmma_f32_16x16x32_bf16(false, a, false, b,
                                                 (short)0, c, false, false);
}

// ---------------------------------------------------------------------------
// bf16 WMMA GEMM:  C[M,N] = A[M,K] (row-major, lda) x W[N,K]^T + bias
// Each wave computes a 32(M) x 64(N) block: 2 A-frags x 4 B-frags -> 8 WMMAs
// per k-step from 12 b128 loads (2x the math per byte of the 16x64 version).
// M % 32 == 0, N % 64 == 0, K % 32 == 0 guaranteed by padding.
// ---------------------------------------------------------------------------
template <bool WF, bool WB>
__global__ void gemm_bf16_wmma(const __bf16* __restrict__ A, int lda,
                               const __bf16* __restrict__ W, int ldw,
                               const float*  __restrict__ bias,
                               float* __restrict__ Cf, __bf16* __restrict__ Cb,
                               int ldc, int K, int mTiles, int nStrips) {
  const int wave = threadIdx.x >> 5;
  const int g    = blockIdx.x * (blockDim.x >> 5) + wave;
  if (g >= mTiles * nStrips) return;   // wave-uniform; before any acc is live
  const int mt = g % mTiles;
  const int ns = g / mTiles;
  const int m0 = mt * 32;
  const int n0 = ns * 64;
  const int lane = threadIdx.x & 31;
  const int hf   = lane >> 4;   // 0: lanes 0-15, 1: lanes 16-31
  const int lm   = lane & 15;

  v8f acc00 = {}, acc01 = {}, acc02 = {}, acc03 = {};
  v8f acc10 = {}, acc11 = {}, acc12 = {}, acc13 = {};

  // A fragment rows: elements 0..7 = K(k0+8*hf ..), 8..15 = K(k0+16+8*hf ..)
  const __bf16* ap0 = A + (size_t)(m0 + lm) * lda + 8 * hf;
  const __bf16* ap1 = ap0 + (size_t)16 * lda;
  // B fragment rows: elements 0..7 = K(k0+8*hf ..), 8..15 = K(k0+16+8*hf ..)   (B = W^T, K-major)
  const __bf16* wp0 = W + (size_t)(n0 + lm) * ldw + 8 * hf;
  const __bf16* wp1 = wp0 + (size_t)16 * ldw;
  const __bf16* wp2 = wp0 + (size_t)32 * ldw;
  const __bf16* wp3 = wp0 + (size_t)48 * ldw;

#pragma unroll 2
  for (int k0 = 0; k0 < K; k0 += 32) {
    const v16bf a0 = load_frag16(ap0 + k0, ap0 + k0 + 16);
    const v16bf a1 = load_frag16(ap1 + k0, ap1 + k0 + 16);
    const v16bf b0 = load_frag16(wp0 + k0, wp0 + k0 + 16);
    acc00 = wmma_bf16(a0, b0, acc00);
    acc10 = wmma_bf16(a1, b0, acc10);
    const v16bf b1 = load_frag16(wp1 + k0, wp1 + k0 + 16);
    acc01 = wmma_bf16(a0, b1, acc01);
    acc11 = wmma_bf16(a1, b1, acc11);
    const v16bf b2 = load_frag16(wp2 + k0, wp2 + k0 + 16);
    acc02 = wmma_bf16(a0, b2, acc02);
    acc12 = wmma_bf16(a1, b2, acc12);
    const v16bf b3 = load_frag16(wp3 + k0, wp3 + k0 + 16);
    acc03 = wmma_bf16(a0, b3, acc03);
    acc13 = wmma_bf16(a1, b3, acc13);
  }

  // Epilogue: C layout: VGPR r -> M = m0 + 16*i + r + 8*hf, N = n0 + 16*t + lm
  v8f* accs[2][4] = {{&acc00, &acc01, &acc02, &acc03},
                     {&acc10, &acc11, &acc12, &acc13}};
#pragma unroll
  for (int i = 0; i < 2; ++i) {
#pragma unroll
    for (int t = 0; t < 4; ++t) {
      const int nn = n0 + t * 16 + lm;
      const float bv = bias[nn];
      const v8f av = *accs[i][t];
#pragma unroll
      for (int r = 0; r < 8; ++r) {
        const int mm = m0 + 16 * i + r + 8 * hf;
        const float v = av[r] + bv;
        if constexpr (WF) Cf[(size_t)mm * ldc + nn] = v;
        if constexpr (WB) Cb[(size_t)mm * ldc + nn] = (__bf16)v;
      }
    }
  }
}

// ---------------------------------------------------------------------------
// fp32 -> bf16 conversion
// ---------------------------------------------------------------------------
__global__ void f2bf_kernel(const float* __restrict__ src, __bf16* __restrict__ dst, int n) {
  int i = blockIdx.x * blockDim.x + threadIdx.x;
  if (i < n) dst[i] = (__bf16)src[i];
}

// Fuse W_ih [4H, E+H] and W_hh [4H, H] into W_cat [4H, XH]; b_cat = b_ih+b_hh
__global__ void build_wcat_kernel(const float* __restrict__ Wih, const float* __restrict__ Whh,
                                  const float* __restrict__ bih, const float* __restrict__ bhh,
                                  __bf16* __restrict__ Wcat, float* __restrict__ bcat) {
  int i = blockIdx.x * blockDim.x + threadIdx.x;
  if (i < G4_ * XH_) {
    int n = i / XH_, k = i % XH_;
    float v = (k < EH_) ? Wih[(size_t)n * EH_ + k] : Whh[(size_t)n * H_ + (k - EH_)];
    Wcat[i] = (__bf16)v;
  }
  if (i < G4_) bcat[i] = bih[i] + bhh[i];
}

// Zero the padded W_out rows [V_..VP_) and build padded bias
__global__ void pad_vocab_kernel(const float* __restrict__ b_out,
                                 __bf16* __restrict__ Wout_b, float* __restrict__ bout_p) {
  int i = blockIdx.x * blockDim.x + threadIdx.x;
  if (i < (VP_ - V_) * H_) Wout_b[(size_t)V_ * H_ + i] = (__bf16)0.0f;
  if (i < VP_) bout_p[i] = (i < V_) ? b_out[i] : 0.0f;
}

// h=0, c=0, cur=BOS(1); zero entire packed xh row
__global__ void init_state_kernel(float* __restrict__ c, __bf16* __restrict__ xh,
                                  int* __restrict__ cur, int B) {
  int i = blockIdx.x * blockDim.x + threadIdx.x;
  if (i < B * XH_) xh[i] = (__bf16)0.0f;
  if (i < B * H_)  c[i] = 0.0f;
  if (i < B)       cur[i] = 1;  // BOS
}

// ---------------------------------------------------------------------------
// Attention: e_r = tanh(f[b,r,:] + kq[b,:]) . W_v + b_v ; softmax ; ctx.
// Also gathers emb row for cur[b] and writes bf16 [emb|ctx] into xh.
// One block per batch row, 256 threads (8 waves).
// ---------------------------------------------------------------------------
__global__ void attention_kernel(const float* __restrict__ f,
                                 const float* __restrict__ kq,
                                 const float* __restrict__ Wv, const float* __restrict__ bv,
                                 const float* __restrict__ embW,
                                 const int* __restrict__ cur,
                                 __bf16* __restrict__ xh) {
  const int b = blockIdx.x;
  const float* fb = f + (size_t)b * R_ * H_;
  const float* kb = kq + (size_t)b * H_;
  const int tid = threadIdx.x, lane = tid & 31, wave = tid >> 5;

  __shared__ float s_e[R_];

  for (int r = wave; r < R_; r += 8) {
    float s = 0.0f;
    for (int h = lane; h < H_; h += 32)
      s += tanhf(fb[r * H_ + h] + kb[h]) * Wv[h];
#pragma unroll
    for (int off = 16; off > 0; off >>= 1) s += __shfl_xor(s, off, 32);
    if (lane == 0) s_e[r] = s + bv[0];
  }
  __syncthreads();

  // softmax over R=36 (cheap, recomputed per thread from LDS)
  float mx = -1e30f;
#pragma unroll
  for (int r = 0; r < R_; ++r) mx = fmaxf(mx, s_e[r]);
  float den = 0.0f;
#pragma unroll
  for (int r = 0; r < R_; ++r) den += __expf(s_e[r] - mx);
  const float inv = 1.0f / den;

  __bf16* xrow = xh + (size_t)b * XH_;

  // embedding gather into xh[0:256]
  if (tid < E_) {
    const int tok = cur[b];
    xrow[tid] = (__bf16)embW[(size_t)tok * E_ + tid];
  }
  // context into xh[256:768]  (each thread: h = tid, tid+256)
  for (int h = tid; h < H_; h += 256) {
    float ctx = 0.0f;
#pragma unroll
    for (int r = 0; r < R_; ++r)
      ctx += (__expf(s_e[r] - mx) * inv) * fb[r * H_ + h];
    xrow[E_ + h] = (__bf16)ctx;
  }
}

// ---------------------------------------------------------------------------
// LSTM pointwise: gates [B,4H] -> c2, h2; h2 written bf16 into xh h-slot
// ---------------------------------------------------------------------------
__device__ __forceinline__ float sigmf(float x) { return 1.0f / (1.0f + __expf(-x)); }

__global__ void lstm_kernel(const float* __restrict__ gates,
                            float* __restrict__ c, __bf16* __restrict__ xh, int B) {
  int idx = blockIdx.x * blockDim.x + threadIdx.x;
  if (idx >= B * H_) return;
  int b = idx >> 9, j = idx & (H_ - 1);
  const float* g = gates + (size_t)b * G4_;
  float ig = sigmf(g[j]);
  float fg = sigmf(g[H_ + j]);
  float gg = tanhf(g[2 * H_ + j]);
  float og = sigmf(g[3 * H_ + j]);
  float c2 = fg * c[idx] + ig * gg;
  float h2 = og * tanhf(c2);
  c[idx] = c2;
  xh[(size_t)b * XH_ + EH_ + j] = (__bf16)h2;
}

// ---------------------------------------------------------------------------
// Per-row argmax over V=10000 logits (row stride VP_); first-index tie-break.
// ---------------------------------------------------------------------------
__global__ void argmax_kernel(const float* __restrict__ logits,
                              int* __restrict__ cur, int* __restrict__ out_tok,
                              int t, int steps) {
  const int b = blockIdx.x;
  const float* lb = logits + (size_t)b * VP_;
  __shared__ float s_v[256];
  __shared__ int   s_i[256];
  float best = -3.4e38f;
  int bi = 0;
  for (int v = threadIdx.x; v < V_; v += 256) {
    float x = lb[v];
    if (x > best) { best = x; bi = v; }   // ascending scan keeps first index
  }
  s_v[threadIdx.x] = best;
  s_i[threadIdx.x] = bi;
  __syncthreads();
  for (int s = 128; s > 0; s >>= 1) {
    if (threadIdx.x < s) {
      float ov = s_v[threadIdx.x + s];
      int   oi = s_i[threadIdx.x + s];
      if (ov > s_v[threadIdx.x] ||
          (ov == s_v[threadIdx.x] && oi < s_i[threadIdx.x])) {
        s_v[threadIdx.x] = ov;
        s_i[threadIdx.x] = oi;
      }
    }
    __syncthreads();
  }
  if (threadIdx.x == 0) {
    cur[b] = s_i[0];
    out_tok[(size_t)b * steps + t] = s_i[0];
  }
}

// ---------------------------------------------------------------------------
// Host side
// ---------------------------------------------------------------------------
enum class OutMode { F32, BF16 };

static inline void launch_gemm(const __bf16* A, int lda, const __bf16* W, int ldw,
                               const float* bias, float* Cf, __bf16* Cb, int ldc,
                               int M, int N, int K, OutMode mode, hipStream_t stream) {
  int mTiles = M / 32;              // 32-row tiles (M always a multiple of 32)
  int nStrips = N / 64;             // N is always a multiple of 64 (padded)
  int waves = mTiles * nStrips;
  int blocks = (waves + 7) / 8;     // 8 waves (256 threads) per block
  if (mode == OutMode::F32) {
    gemm_bf16_wmma<true, false><<<blocks, 256, 0, stream>>>(
        A, lda, W, ldw, bias, Cf, Cb, ldc, K, mTiles, nStrips);
  } else {
    gemm_bf16_wmma<false, true><<<blocks, 256, 0, stream>>>(
        A, lda, W, ldw, bias, Cf, Cb, ldc, K, mTiles, nStrips);
  }
}

extern "C" void kernel_launch(void* const* d_in, const int* in_sizes, int n_in,
                              void* d_out, int out_size, void* d_ws, size_t ws_size,
                              hipStream_t stream) {
  const float* feats = (const float*)d_in[0];
  const float* embW  = (const float*)d_in[1];
  const float* W_fp  = (const float*)d_in[2];
  const float* b_fp  = (const float*)d_in[3];
  const float* W_q   = (const float*)d_in[4];
  const float* b_q   = (const float*)d_in[5];
  const float* W_k   = (const float*)d_in[6];
  const float* b_k   = (const float*)d_in[7];
  const float* W_v   = (const float*)d_in[8];
  const float* b_v   = (const float*)d_in[9];
  const float* W_ih  = (const float*)d_in[10];
  const float* b_ih  = (const float*)d_in[11];
  const float* W_hh  = (const float*)d_in[12];
  const float* b_hh  = (const float*)d_in[13];
  const float* W_out = (const float*)d_in[14];
  const float* b_out = (const float*)d_in[15];
  // d_in[16] = max_len (device scalar) -- cannot sync-read during graph
  // capture; derive step count from host-visible out_size instead.

  const int B     = in_sizes[0] / (R_ * D_);   // 128
  const int steps = out_size / B;              // max_len - 1 = 29
  int* out_tok = (int*)d_out;

  // ---- workspace carve (256B aligned) ----
  char* p = (char*)d_ws;
  auto carve = [&](size_t bytes) -> void* {
    void* r = (void*)p;
    p += (bytes + 255) & ~(size_t)255;
    return r;
  };
  __bf16* feats_b = (__bf16*)carve((size_t)B * R_ * D_ * 2);
  __bf16* Wfp_b   = (__bf16*)carve((size_t)H_ * D_ * 2);
  __bf16* Wq_b    = (__bf16*)carve((size_t)H_ * H_ * 2);
  __bf16* Wk_b    = (__bf16*)carve((size_t)H_ * H_ * 2);
  __bf16* Wcat_b  = (__bf16*)carve((size_t)G4_ * XH_ * 2);
  float*  bcat    = (float*)carve((size_t)G4_ * 4);
  __bf16* Wout_b  = (__bf16*)carve((size_t)VP_ * H_ * 2);  // padded vocab rows
  float*  bout_p  = (float*)carve((size_t)VP_ * 4);        // padded bias
  float*  f_f32   = (float*)carve((size_t)B * R_ * H_ * 4);
  __bf16* xh      = (__bf16*)carve((size_t)B * XH_ * 2);   // [emb|ctx|h] bf16
  __bf16* q_b     = (__bf16*)carve((size_t)B * H_ * 2);
  float*  kq_f    = (float*)carve((size_t)B * H_ * 4);
  float*  gates   = (float*)carve((size_t)B * G4_ * 4);
  float*  logits  = (float*)carve((size_t)B * VP_ * 4);    // padded stride
  float*  c_st    = (float*)carve((size_t)B * H_ * 4);
  int*    cur     = (int*)carve((size_t)B * 4);

  auto cvt = [&](const float* s, __bf16* d, int n) {
    f2bf_kernel<<<(n + 255) / 256, 256, 0, stream>>>(s, d, n);
  };

  // ---- one-time conversions ----
  cvt(feats, feats_b, B * R_ * D_);
  cvt(W_fp, Wfp_b, H_ * D_);
  cvt(W_q, Wq_b, H_ * H_);
  cvt(W_k, Wk_b, H_ * H_);
  cvt(W_out, Wout_b, V_ * H_);
  pad_vocab_kernel<<<((VP_ - V_) * H_ + 255) / 256, 256, 0, stream>>>(
      b_out, Wout_b, bout_p);
  build_wcat_kernel<<<(G4_ * XH_ + 255) / 256, 256, 0, stream>>>(
      W_ih, W_hh, b_ih, b_hh, Wcat_b, bcat);

  // ---- feat projection: f[B*R, H] = feats[B*R, D] @ W_fp^T + b_fp ----
  launch_gemm(feats_b, D_, Wfp_b, D_, b_fp, f_f32, nullptr, H_,
              B * R_, H_, D_, OutMode::F32, stream);

  // ---- init h=c=0, cur=BOS ----
  init_state_kernel<<<(B * XH_ + 255) / 256, 256, 0, stream>>>(c_st, xh, cur, B);

  // ---- serial decode loop ----
  for (int t = 0; t < steps; ++t) {
    // q = h @ W_q^T + b_q          (A = xh h-slot, bf16 out)
    launch_gemm(xh + EH_, XH_, Wq_b, H_, b_q, nullptr, q_b, H_,
                B, H_, H_, OutMode::BF16, stream);
    // kq = q @ W_k^T + b_k         (f32 out)
    launch_gemm(q_b, H_, Wk_b, H_, b_k, kq_f, nullptr, H_,
                B, H_, H_, OutMode::F32, stream);
    // attention + emb gather -> xh [emb|ctx]
    attention_kernel<<<B, 256, 0, stream>>>(f_f32, kq_f, W_v, b_v, embW, cur, xh);
    // gates = [emb|ctx|h] @ [W_ih|W_hh]^T + (b_ih + b_hh)
    launch_gemm(xh, XH_, Wcat_b, XH_, bcat, gates, nullptr, G4_,
                B, G4_, XH_, OutMode::F32, stream);
    // LSTM cell: c,h update; h2 -> xh h-slot (bf16)
    lstm_kernel<<<(B * H_ + 255) / 256, 256, 0, stream>>>(gates, c_st, xh, B);
    // logits = h2 @ W_out^T + b_out (padded N)
    launch_gemm(xh + EH_, XH_, Wout_b, H_, bout_p, logits, nullptr, VP_,
                B, VP_, H_, OutMode::F32, stream);
    // argmax -> cur, token
    argmax_kernel<<<B, 256, 0, stream>>>(logits, cur, out_tok, t, steps);
  }
}
